// VectorQuantizerPT_21869973471295
// MI455X (gfx1250) — matrix operand order repaired and checked by reading, not compiled, not run on hardware
//
#include <hip/hip_runtime.h>

#define N_ROWS   65536
#define DIM      256
#define K_CODES  1024
#define ROWS_PER_BLOCK 32
#define DIST_PITCH 1028      // 1024 + 4 pad floats: kills LDS bank conflicts on WMMA-layout stores

typedef __attribute__((ext_vector_type(16))) __bf16 v16bf;
typedef __attribute__((ext_vector_type(8)))  float  v8f;

__device__ __forceinline__ void split_bf16(float f, __bf16& h, __bf16& l) {
  h = (__bf16)f;                 // RNE convert to bf16 (top ~8 mantissa bits)
  l = (__bf16)(f - (float)h);    // residual (next ~8 bits) -> 3xBF16 ~ fp32 GEMM
}

// ---------------- prep kernels (codebook-side, run once per launch) ----------------

// cnorm[k] = sum_d codebook[d][k]^2 ; also zero the loss accumulator slot.
__global__ void vq_prep_norm(const float* __restrict__ cb, float* __restrict__ cnorm,
                             float* __restrict__ loss_out) {
  int k = blockIdx.x * blockDim.x + threadIdx.x;          // 1024 threads total, coalesced over k
  float s = 0.f;
  for (int d = 0; d < DIM; ++d) { float v = cb[d * K_CODES + k]; s += v * v; }
  cnorm[k] = s;
  if (k == 0) loss_out[0] = 0.f;
}

// cbT[k][d] = codebook[d][k]  (for coalesced quantized gathers)
__global__ void vq_prep_transpose(const float* __restrict__ cb, float* __restrict__ cbT) {
  int k = blockIdx.x;      // 1024
  int d = threadIdx.x;     // 256
  cbT[k * DIM + d] = cb[d * K_CODES + k];
}

// Pack codebook into fragment-major bf16 hi/lo buffers matching the CDNA5
// 16-bit B (32x16) VGPR layout: lane L holds column (L&15); element j holds
// inner-dim row K = (j>>3)*16 + (L>>4)*8 + (j&7). Main kernel then loads one contiguous
// 32B v16bf per lane per (tile, k-step).
__global__ void vq_prep_bfrag(const float* __restrict__ cb,
                              __bf16* __restrict__ Bhi, __bf16* __restrict__ Blo) {
  int ct = blockIdx.x >> 3;          // column tile 0..63
  int s  = blockIdx.x & 7;           // k-step 0..7 (32 inner dims each)
  int t  = threadIdx.x;              // 512 threads: L = t>>4 (lane), j = t&15 (element)
  int L  = t >> 4;
  int j  = t & 15;
  int col = (ct << 4) + (L & 15);
  int d   = (s << 5) + ((j >> 3) << 4) + ((L >> 4) << 3) + (j & 7);
  float v = cb[d * K_CODES + col];
  __bf16 h, l; split_bf16(v, h, l);
  size_t off = (size_t)blockIdx.x * 512 + t;   // ((ct*8+s)*32 + L)*16 + j
  Bhi[off] = h; Blo[off] = l;
}

// ---------------- main kernel: 1 block = 32 rows x 1024 codes ----------------
// 8 waves; wave w owns col-tiles w*8..w*8+7, processed in 2 groups of 4.
// Each loaded B fragment feeds BOTH row-tiles (6 WMMAs) -> halves B L2 traffic.

__global__ __launch_bounds__(256)
void vq_main(const float* __restrict__ x, const float* __restrict__ cbT,
             const float* __restrict__ cnorm,
             const __bf16* __restrict__ Bhi, const __bf16* __restrict__ Blo,
             float* __restrict__ quant_out, float* __restrict__ soft_out,
             float* __restrict__ loss_out) {
  extern __shared__ float smem[];
  float* sx    = smem;                          // 32*256   x tile (fp32)       32KB
  float* sdist = sx + ROWS_PER_BLOCK * DIM;     // 32*1028  dist tile (padded) 128.5KB
  float* sxn   = sdist + ROWS_PER_BLOCK * DIST_PITCH; // 32  row |x|^2
  float* rmin  = sxn + ROWS_PER_BLOCK;          // 256 reduction scratch
  float* rsum  = rmin + 256;                    // 256
  int*   ridx  = (int*)(rsum + 256);            // 256
  float* s_rowsuminv = (float*)(ridx + 256);    // 32
  int*   s_rowidx    = (int*)(s_rowsuminv + ROWS_PER_BLOCK); // 32

  const int tid     = threadIdx.x;
  const int lane    = tid & 31;
  const int wave    = tid >> 5;
  const int halfsel = lane >> 4;       // which 16-lane half
  const int mrow    = lane & 15;
  const int aoff    = halfsel * 8;     // A-frag K offset / C-matrix row offset
  const int rowbase = blockIdx.x * ROWS_PER_BLOCK;

  // ---- phase 1: stage x tile (32x256 fp32 = 32KB) coalesced, float4 ----
  {
    const float4* xg  = (const float4*)(x + (size_t)rowbase * DIM);
    float4*       sx4 = (float4*)sx;
#pragma unroll
    for (int i = 0; i < 8; ++i) sx4[tid + i * 256] = xg[tid + i * 256];
  }
  __syncthreads();

  // ---- phase 2: per-row |x|^2 (8 threads/row partials, then reduce) ----
  {
    int row = tid >> 3, seg = (tid & 7) << 5;
    const float* rp = sx + row * DIM + seg;
    float s = 0.f;
#pragma unroll
    for (int j = 0; j < 32; ++j) { float v = rp[j]; s += v * v; }
    rmin[tid] = s;
  }
  __syncthreads();
  if (tid < ROWS_PER_BLOCK) {
    float s = 0.f;
    for (int j = 0; j < 8; ++j) s += rmin[tid * 8 + j];
    sxn[tid] = s;
  }
  __syncthreads();

  // ---- phase 3: 3xBF16 WMMA GEMM, 2 row-tiles x (2 groups of 4 col-tiles) ----
  float xnr[2][8];
#pragma unroll
  for (int rt = 0; rt < 2; ++rt)
#pragma unroll
    for (int r = 0; r < 8; ++r) xnr[rt][r] = sxn[rt * 16 + r + aoff];

  for (int g = 0; g < 2; ++g) {
    v8f acc[2][4];
    {
      v8f z = {};
#pragma unroll
      for (int rt = 0; rt < 2; ++rt)
#pragma unroll
        for (int t = 0; t < 4; ++t) acc[rt][t] = z;
    }

    for (int s = 0; s < 8; ++s) {        // 8 k-steps of 32 over DIM=256
      // Build A fragments for both row-tiles (ISA 16-bit A layout:
      // half-lane K offsets {0..7,16..23} / {8..15,24..31})
      v16bf ahi[2], alo[2];
#pragma unroll
      for (int rt = 0; rt < 2; ++rt) {
        const float* ap = sx + (rt * 16 + mrow) * DIM + s * 32 + aoff;
        float4 a0 = *(const float4*)(ap);
        float4 a1 = *(const float4*)(ap + 4);
        float4 a2 = *(const float4*)(ap + 16);
        float4 a3 = *(const float4*)(ap + 20);
        float av[16] = { a0.x, a0.y, a0.z, a0.w, a1.x, a1.y, a1.z, a1.w,
                         a2.x, a2.y, a2.z, a2.w, a3.x, a3.y, a3.z, a3.w };
#pragma unroll
        for (int j = 0; j < 16; ++j) {
          __bf16 h, l; split_bf16(av[j], h, l);
          ahi[rt][j] = h; alo[rt][j] = l;
        }
      }

#pragma unroll
      for (int t = 0; t < 4; ++t) {
        int ct = wave * 8 + g * 4 + t;
        size_t fo = ((size_t)((ct << 3) + s) << 9) + ((size_t)lane << 4);
        v16bf bh = *(const v16bf*)(Bhi + fo);
        v16bf bl = *(const v16bf*)(Blo + fo);
        // sim ~= Ahi*Bhi + Ahi*Blo + Alo*Bhi (fp32 acc), B reused by both row-tiles
#pragma unroll
        for (int rt = 0; rt < 2; ++rt) {
          acc[rt][t] = __builtin_amdgcn_wmma_f32_16x16x32_bf16(false, ahi[rt], false, bh, (short)0, acc[rt][t], false, false);
          acc[rt][t] = __builtin_amdgcn_wmma_f32_16x16x32_bf16(false, ahi[rt], false, bl, (short)0, acc[rt][t], false, false);
          acc[rt][t] = __builtin_amdgcn_wmma_f32_16x16x32_bf16(false, alo[rt], false, bh, (short)0, acc[rt][t], false, false);
        }
      }
    }

    // dist = |x|^2 + |c|^2 - 2*sim -> padded LDS tile
#pragma unroll
    for (int t = 0; t < 4; ++t) {
      int n  = ((wave * 8 + g * 4 + t) << 4) + mrow;
      float cn = cnorm[n];
#pragma unroll
      for (int rt = 0; rt < 2; ++rt) {
#pragma unroll
        for (int r = 0; r < 8; ++r) {
          float dst = xnr[rt][r] + cn - 2.0f * acc[rt][t][r];
          sdist[(rt * 16 + r + aoff) * DIST_PITCH + n] = dst;
        }
      }
    }
  }
  __syncthreads();

  // ---- phase 4: per-row argmin + sum (1/d)^2 ; 8 threads per row ----
  {
    int row = tid >> 3, c0 = tid & 7;
    const float* dr = sdist + row * DIST_PITCH;
    float best = 3.4e38f; int bidx = 0; float ssum = 0.f;
    for (int c = c0; c < K_CODES; c += 8) {
      float d = dr[c];
      float inv = 1.0f / d;
      ssum += inv * inv;
      if (d < best || (d == best && c < bidx)) { best = d; bidx = c; }
    }
    rmin[tid] = best; ridx[tid] = bidx; rsum[tid] = ssum;
  }
  __syncthreads();
  if (tid < ROWS_PER_BLOCK) {
    float best = 3.4e38f; int bidx = 0; float ssum = 0.f;
    for (int j = 0; j < 8; ++j) {
      int p = tid * 8 + j;
      float d = rmin[p]; int ix = ridx[p];
      if (d < best || (d == best && ix < bidx)) { best = d; bidx = ix; }
      ssum += rsum[p];
    }
    s_rowidx[tid]    = bidx;
    s_rowsuminv[tid] = 1.0f / ssum;
  }
  __syncthreads();

  // ---- phase 5: soft_counts = (1/d)^2 / rowsum, coalesced 256-wide stores ----
  for (int r = 0; r < ROWS_PER_BLOCK; ++r) {
    float sinv = s_rowsuminv[r];
    const float* dr = sdist + r * DIST_PITCH;
    float* gp = soft_out + (size_t)(rowbase + r) * K_CODES;
#pragma unroll
    for (int i = 0; i < 4; ++i) {
      int c = tid + i * 256;
      float d = dr[c];
      float inv = 1.0f / d;
      gp[c] = inv * inv * sinv;
    }
  }

  // ---- phase 6: quantized gather (cbT row, coalesced) + fused loss ----
  {
    int row = tid >> 3;
    int dd  = (tid & 7) << 5;
    int k   = s_rowidx[row];
    const float4* cp = (const float4*)(cbT + (size_t)k * DIM + dd);
    const float4* xp = (const float4*)(sx + row * DIM + dd);
    float4*       qp = (float4*)(quant_out + (size_t)(rowbase + row) * DIM + dd);
    float lsum = 0.f;
#pragma unroll
    for (int j = 0; j < 8; ++j) {
      float4 q = cp[j], xv = xp[j];
      qp[j] = q;
      float dx = q.x - xv.x, dy = q.y - xv.y, dz = q.z - xv.z, dw = q.w - xv.w;
      lsum += dx * dx + dy * dy + dz * dz + dw * dw;
    }
    rmin[tid] = lsum;
  }
  __syncthreads();
  if (tid == 0) {
    float tot = 0.f;
    for (int j = 0; j < 256; ++j) tot += rmin[j];
    // vq_loss = (beta + 1) * mean((q - x)^2) = 1.25 * sum / (N*D)
    atomicAdd(loss_out, tot * (1.25f / 16777216.0f));
  }
}

// ---------------- host entry ----------------

extern "C" void kernel_launch(void* const* d_in, const int* in_sizes, int n_in,
                              void* d_out, int out_size, void* d_ws, size_t ws_size,
                              hipStream_t stream) {
  (void)in_sizes; (void)n_in; (void)out_size; (void)ws_size;
  const float* x  = (const float*)d_in[0];
  const float* cb = (const float*)d_in[1];

  float* out       = (float*)d_out;
  float* quant_out = out;                                        // 65536*256
  float* soft_out  = out + (size_t)N_ROWS * DIM;                 // 65536*1024
  float* loss_out  = out + (size_t)N_ROWS * DIM + (size_t)N_ROWS * K_CODES;

  // workspace: cnorm(4KB) | cbT(1MB) | Bhi frags(512KB) | Blo frags(512KB)
  float*  cnorm = (float*)d_ws;
  float*  cbT   = cnorm + K_CODES;
  __bf16* Bhi   = (__bf16*)(cbT + (size_t)K_CODES * DIM);
  __bf16* Blo   = Bhi + (size_t)K_CODES * DIM;

  vq_prep_norm<<<K_CODES / 256, 256, 0, stream>>>(cb, cnorm, loss_out);
  vq_prep_transpose<<<K_CODES, DIM, 0, stream>>>(cb, cbT);
  vq_prep_bfrag<<<512, 512, 0, stream>>>(cb, Bhi, Blo);

  const int SMEM_BYTES =
      (ROWS_PER_BLOCK * DIM + ROWS_PER_BLOCK * DIST_PITCH + ROWS_PER_BLOCK +
       256 + 256 + 256 + ROWS_PER_BLOCK + ROWS_PER_BLOCK) * 4;   // ~168KB, < 320KB/WGP
  (void)hipFuncSetAttribute((const void*)vq_main,
                            hipFuncAttributeMaxDynamicSharedMemorySize, SMEM_BYTES);
  vq_main<<<N_ROWS / ROWS_PER_BLOCK, 256, SMEM_BYTES, stream>>>(
      x, cbT, cnorm, Bhi, Blo, quant_out, soft_out, loss_out);
}
